// Transformer_6794638262339
// MI455X (gfx1250) — matrix operand order repaired and checked by reading, not compiled, not run on hardware
//
#include <hip/hip_runtime.h>
#include <hip/hip_bf16.h>
#include <math.h>

// ---------------------------------------------------------------------------
// Types for CDNA5 WMMA (wave32, 16x16x32 bf16 -> f32)
// ---------------------------------------------------------------------------
typedef __bf16 v16bf __attribute__((ext_vector_type(16)));
typedef float  v8f   __attribute__((ext_vector_type(8)));

struct __align__(16) U4 { unsigned int x, y, z, w; };
union FragBF { v16bf v; U4 q[2]; };

__device__ __forceinline__ unsigned short f32_to_bf16(float f) {
    unsigned int u = __float_as_uint(f);
    u += 0x7FFFu + ((u >> 16) & 1u);          // round-to-nearest-even
    return (unsigned short)(u >> 16);
}
__device__ __forceinline__ float bf16_to_f32(unsigned short h) {
    return __uint_as_float(((unsigned int)h) << 16);
}

#define B_   8
#define N_   1024
#define DIM_ 768
#define HEADS_ 12
#define DHEAD_ 64
#define INNER_ 768
#define MLP_ 3072
#define M_TOT (B_ * N_)              // 8192 rows
#define QKV_LD (3 * INNER_)          // 2304
#define SCALE_ 0.125f                // 64^-0.5

// ---------------------------------------------------------------------------
// Ternary weight quantization (deterministic 2-stage reduction)
// ---------------------------------------------------------------------------
__global__ __launch_bounds__(256) void k_reduce_abs(const float* __restrict__ w,
                                                    long n, float* __restrict__ partials) {
    __shared__ float red[256];
    float s = 0.f;
    for (long i = (long)blockIdx.x * 256 + threadIdx.x; i < n; i += 256L * 256L)
        s += fabsf(w[i]);
    red[threadIdx.x] = s; __syncthreads();
    for (int off = 128; off; off >>= 1) {
        if ((int)threadIdx.x < off) red[threadIdx.x] += red[threadIdx.x + off];
        __syncthreads();
    }
    if (threadIdx.x == 0) partials[blockIdx.x] = red[0];
}

__global__ __launch_bounds__(256) void k_finalize_scale(const float* __restrict__ partials,
                                                        long n, float* __restrict__ scale) {
    __shared__ float red[256];
    red[threadIdx.x] = partials[threadIdx.x]; __syncthreads();
    for (int off = 128; off; off >>= 1) {
        if ((int)threadIdx.x < off) red[threadIdx.x] += red[threadIdx.x + off];
        __syncthreads();
    }
    if (threadIdx.x == 0) scale[0] = red[0] / (float)n + 1e-8f;
}

__global__ __launch_bounds__(256) void k_quant(const float* __restrict__ w, long n,
                                               const float* __restrict__ scale,
                                               unsigned short* __restrict__ wq, int ternary) {
    float s = scale[0];
    for (long i = (long)blockIdx.x * 256 + threadIdx.x; i < n; i += (long)gridDim.x * 256L) {
        float v = w[i];
        if (ternary) {
            float q = rintf(v / s);                      // jnp.round == RNE
            q = fminf(1.f, fmaxf(-1.f, q));
            v = q * s;
        }
        wq[i] = f32_to_bf16(v);
    }
}

// ---------------------------------------------------------------------------
// LayerNorm: one block per row, input f32 OR bf16, output bf16
// ---------------------------------------------------------------------------
__global__ __launch_bounds__(256) void k_layernorm(const float* __restrict__ inf,
                                                   const unsigned short* __restrict__ inb,
                                                   const float* __restrict__ gam,
                                                   const float* __restrict__ bet,
                                                   unsigned short* __restrict__ out, int F) {
    __shared__ float red[256];
    const int tid = threadIdx.x;
    const int nper = F >> 8;                             // 3 (768) or 12 (3072)
    const size_t base = (size_t)blockIdx.x * F;
    float x[12];
    float s = 0.f;
    for (int i = 0; i < nper; ++i) {
        int idx = tid + (i << 8);
        float v = inf ? inf[base + idx] : bf16_to_f32(inb[base + idx]);
        x[i] = v; s += v;
    }
    red[tid] = s; __syncthreads();
    for (int off = 128; off; off >>= 1) {
        if (tid < off) red[tid] += red[tid + off];
        __syncthreads();
    }
    float mean = red[0] / (float)F;
    __syncthreads();
    float vs = 0.f;
    for (int i = 0; i < nper; ++i) { float d = x[i] - mean; vs += d * d; }
    red[tid] = vs; __syncthreads();
    for (int off = 128; off; off >>= 1) {
        if (tid < off) red[tid] += red[tid + off];
        __syncthreads();
    }
    float rstd = rsqrtf(red[0] / (float)F + 1e-5f);
    for (int i = 0; i < nper; ++i) {
        int idx = tid + (i << 8);
        out[base + idx] = f32_to_bf16((x[i] - mean) * rstd * gam[idx] + bet[idx]);
    }
}

// ---------------------------------------------------------------------------
// GEMM: C[M,Nout] = A[M,K](bf16) * W[Nout,K]^T(bf16)
//   mode 0: out_bf16 = acc (+bias)        (qkv)
//   mode 1: out_bf16 = gelu(acc + bias)   (W1)
//   mode 2: h_f32   += acc + bias         (Wo, W2 residual)
// Block: 256 thr = 8 waves, tile 128x128, K-step 32.
// Double-buffered LDS (40KB): next tile's global b128 loads are issued before
// the current tile's WMMAs, stored to LDS after them; 1 barrier per K-step.
// Wave (waveM 0..1, waveN 0..3) owns 64x32 => 4x2 WMMA accumulators.
// ---------------------------------------------------------------------------
__global__ __launch_bounds__(256) void k_gemm_bf16(const unsigned short* __restrict__ A, int lda,
                                                   const unsigned short* __restrict__ W, int ldw,
                                                   const float* __restrict__ bias,
                                                   unsigned short* __restrict__ outb, int ldo,
                                                   float* __restrict__ hacc,
                                                   int K, int mode) {
    __shared__ __align__(16) unsigned short Alds[2][128 * 40];  // pad stride 40 (80B, 16B aligned)
    __shared__ __align__(16) unsigned short Wlds[2][128 * 40];
    const int tid = threadIdx.x;
    const int wid = tid >> 5;
    const int lane = tid & 31;
    const int g = lane >> 4;
    const int l16 = lane & 15;
    const int m0 = blockIdx.x * 128;
    const int n0 = blockIdx.y * 128;
    const int waveM = wid >> 2;       // 0..1
    const int waveN = wid & 3;        // 0..3

    // copy assignment: thread -> rows (tid>>2) and (tid>>2)+64, 8-bf16 chunk (tid&3)
    const int row0 = tid >> 2;
    const int row1 = row0 + 64;
    const int ch8 = (tid & 3) * 8;
    const unsigned short* Ap0 = &A[(size_t)(m0 + row0) * lda + ch8];
    const unsigned short* Ap1 = &A[(size_t)(m0 + row1) * lda + ch8];
    const unsigned short* Wp0 = &W[(size_t)(n0 + row0) * ldw + ch8];
    const unsigned short* Wp1 = &W[(size_t)(n0 + row1) * ldw + ch8];

    v8f zero = {};
    v8f acc[4][2];
#pragma unroll
    for (int i = 0; i < 4; ++i)
#pragma unroll
        for (int j = 0; j < 2; ++j) acc[i][j] = zero;

    // prologue: tile 0 -> regs -> LDS buffer 0
    U4 ar0 = *(const U4*)Ap0;
    U4 ar1 = *(const U4*)Ap1;
    U4 wr0 = *(const U4*)Wp0;
    U4 wr1 = *(const U4*)Wp1;
    *(U4*)&Alds[0][row0 * 40 + ch8] = ar0;
    *(U4*)&Alds[0][row1 * 40 + ch8] = ar1;
    *(U4*)&Wlds[0][row0 * 40 + ch8] = wr0;
    *(U4*)&Wlds[0][row1 * 40 + ch8] = wr1;

    const int nk = K >> 5;
    for (int ki = 0; ki < nk; ++ki) {
        const int cur = ki & 1;
        const bool has_next = (ki + 1) < nk;
        // issue next tile's global loads early (4 b128 in flight over the WMMAs)
        if (has_next) {
            const int ko = (ki + 1) << 5;
            ar0 = *(const U4*)&Ap0[ko];
            ar1 = *(const U4*)&Ap1[ko];
            wr0 = *(const U4*)&Wp0[ko];
            wr1 = *(const U4*)&Wp1[ko];
        }
        __syncthreads();   // buffer `cur` fully written; prior reads of `cur^1` done

        FragBF af[4], bfr[2];
#pragma unroll
        for (int mf = 0; mf < 4; ++mf) {
            int r = waveM * 64 + mf * 16 + l16;     // A: lane=row, e0..7 -> k=g*8.., e8..15 -> k=16+g*8..
            af[mf].q[0] = *(const U4*)&Alds[cur][r * 40 + g * 8];
            af[mf].q[1] = *(const U4*)&Alds[cur][r * 40 + 16 + g * 8];
        }
#pragma unroll
        for (int nf = 0; nf < 2; ++nf) {
            int c = waveN * 32 + nf * 16 + l16;     // B: lane=col, e0..7 -> k=g*8.., e8..15 -> k=16+g*8..
            bfr[nf].q[0] = *(const U4*)&Wlds[cur][c * 40 + g * 8];
            bfr[nf].q[1] = *(const U4*)&Wlds[cur][c * 40 + 16 + g * 8];
        }
#pragma unroll
        for (int mf = 0; mf < 4; ++mf)
#pragma unroll
            for (int nf = 0; nf < 2; ++nf)
                acc[mf][nf] = __builtin_amdgcn_wmma_f32_16x16x32_bf16(
                    false, af[mf].v, false, bfr[nf].v, (short)0, acc[mf][nf], false, false);

        if (has_next) {
            const int nxt = cur ^ 1;
            *(U4*)&Alds[nxt][row0 * 40 + ch8] = ar0;
            *(U4*)&Alds[nxt][row1 * 40 + ch8] = ar1;
            *(U4*)&Wlds[nxt][row0 * 40 + ch8] = wr0;
            *(U4*)&Wlds[nxt][row1 * 40 + ch8] = wr1;
        }
    }

    // epilogue: C layout lane=col(l16), row = r + 8*g
#pragma unroll
    for (int mf = 0; mf < 4; ++mf) {
#pragma unroll
        for (int nf = 0; nf < 2; ++nf) {
            int n = n0 + waveN * 32 + nf * 16 + l16;
            float bv = bias ? bias[n] : 0.f;
#pragma unroll
            for (int r = 0; r < 8; ++r) {
                int m = m0 + waveM * 64 + mf * 16 + r + 8 * g;
                float v = acc[mf][nf][r] + bv;
                if (mode == 1) v = 0.5f * v * (1.f + erff(v * 0.70710678118654752f));
                if (mode == 2) hacc[(size_t)m * DIM_ + n] += v;
                else           outb[(size_t)m * ldo + n] = f32_to_bf16(v);
            }
        }
    }
}

// ---------------------------------------------------------------------------
// Flash attention: one wave per (b, head, 16-row Q tile); 8 waves per block.
// qkv: [8192, 2304] bf16 (q|k|v each 768 wide); o: [8192, 768] bf16
// ---------------------------------------------------------------------------
__global__ __launch_bounds__(256) void k_attention(const unsigned short* __restrict__ qkv,
                                                   unsigned short* __restrict__ o) {
    __shared__ __align__(16) unsigned short Vt[8][64 * 40];   // per-wave V^T tile (d-major)
    __shared__ __align__(16) unsigned short Pl[8][16 * 40];   // per-wave P tile (row-major)
    const int tid = threadIdx.x;
    const int wid = tid >> 5;
    const int lane = tid & 31;
    const int g = lane >> 4;
    const int l16 = lane & 15;

    const int job = blockIdx.x * 8 + wid;      // 6144 jobs total
    const int b = job / (HEADS_ * 64);
    const int rem = job % (HEADS_ * 64);
    const int hh = rem >> 6;
    const int qt = rem & 63;
    const int hcol = hh * DHEAD_;
    const int qrow0 = b * N_ + qt * 16;

    // Q fragments (2 chunks of K=32 over head dim)
    FragBF qf[2];
#pragma unroll
    for (int dc = 0; dc < 2; ++dc) {
        const unsigned short* p = &qkv[(size_t)(qrow0 + l16) * QKV_LD + hcol + dc * 32];
        qf[dc].q[0] = *(const U4*)&p[g * 8];
        qf[dc].q[1] = *(const U4*)&p[16 + g * 8];
    }

    v8f zero = {};
    v8f accd[4];
#pragma unroll
    for (int df = 0; df < 4; ++df) accd[df] = zero;
    float mrun[8], lrun[8];
#pragma unroll
    for (int r = 0; r < 8; ++r) { mrun[r] = -1e30f; lrun[r] = 0.f; }

    for (int j0 = 0; j0 < N_; j0 += 32) {
        // S = Q K^T for 32 keys (two 16-key column tiles)
        v8f s[2]; s[0] = zero; s[1] = zero;
#pragma unroll
        for (int jt = 0; jt < 2; ++jt)
#pragma unroll
            for (int dc = 0; dc < 2; ++dc) {
                FragBF kf;  // B frag: lane=key col, e0..7 -> d = dc*32 + g*8.., e8..15 -> dc*32 + 16 + g*8..
                const unsigned short* p =
                    &qkv[(size_t)(b * N_ + j0 + jt * 16 + l16) * QKV_LD + INNER_ + hcol + dc * 32 + g * 8];
                kf.q[0] = *(const U4*)&p[0];
                kf.q[1] = *(const U4*)&p[16];
                s[jt] = __builtin_amdgcn_wmma_f32_16x16x32_bf16(
                    false, qf[dc].v, false, kf.v, (short)0, s[jt], false, false);
            }

        // V rows for this tile: lane <- row (j0+lane), 64 bf16 contiguous
        U4 vv[8];
        const unsigned short* pv = &qkv[(size_t)(b * N_ + j0 + lane) * QKV_LD + 2 * INNER_ + hcol];
#pragma unroll
        for (int c = 0; c < 8; ++c) vv[c] = *(const U4*)&pv[c * 8];

        // online softmax; rows live across lanes (row = r + 8g, col = l16)
        float alpha[8];
#pragma unroll
        for (int r = 0; r < 8; ++r) {
            float s0 = s[0][r] * SCALE_, s1 = s[1][r] * SCALE_;
            float rm = fmaxf(s0, s1);
            rm = fmaxf(rm, __shfl_xor(rm, 1, 32));
            rm = fmaxf(rm, __shfl_xor(rm, 2, 32));
            rm = fmaxf(rm, __shfl_xor(rm, 4, 32));
            rm = fmaxf(rm, __shfl_xor(rm, 8, 32));
            float mnew = fmaxf(mrun[r], rm);
            float p0 = __expf(s0 - mnew), p1 = __expf(s1 - mnew);
            float rs = p0 + p1;
            rs += __shfl_xor(rs, 1, 32);
            rs += __shfl_xor(rs, 2, 32);
            rs += __shfl_xor(rs, 4, 32);
            rs += __shfl_xor(rs, 8, 32);
            alpha[r] = __expf(mrun[r] - mnew);
            lrun[r] = lrun[r] * alpha[r] + rs;
            mrun[r] = mnew;
            Pl[wid][(r + 8 * g) * 40 + l16]      = f32_to_bf16(p0);
            Pl[wid][(r + 8 * g) * 40 + 16 + l16] = f32_to_bf16(p1);
        }
#pragma unroll
        for (int df = 0; df < 4; ++df)
#pragma unroll
            for (int r = 0; r < 8; ++r) accd[df][r] *= alpha[r];

        // transpose V into LDS: Vt[d][j_local]
        const unsigned short* pvs = (const unsigned short*)vv;
#pragma unroll
        for (int d = 0; d < 64; ++d) Vt[wid][d * 40 + lane] = pvs[d];
        __syncthreads();   // uniform across all 8 waves (same trip count)

        // P as A-fragment (16x32), V^T columns as B-fragments
        FragBF pf;
        pf.q[0] = *(const U4*)&Pl[wid][l16 * 40 + g * 8];
        pf.q[1] = *(const U4*)&Pl[wid][l16 * 40 + 16 + g * 8];
#pragma unroll
        for (int df = 0; df < 4; ++df) {
            FragBF vf;  // lane=col d = df*16+l16, e0..7 -> k = g*8.., e8..15 -> k = 16+g*8.. (Vt row)
            vf.q[0] = *(const U4*)&Vt[wid][(df * 16 + l16) * 40 + g * 8];
            vf.q[1] = *(const U4*)&Vt[wid][(df * 16 + l16) * 40 + 16 + g * 8];
            accd[df] = __builtin_amdgcn_wmma_f32_16x16x32_bf16(
                false, pf.v, false, vf.v, (short)0, accd[df], false, false);
        }
        __syncthreads();
    }

    // write O (normalize by running sum)
#pragma unroll
    for (int df = 0; df < 4; ++df)
#pragma unroll
        for (int r = 0; r < 8; ++r) {
            size_t mrow = (size_t)(qrow0 + r + 8 * g);
            o[mrow * INNER_ + hcol + df * 16 + l16] = f32_to_bf16(accd[df][r] / lrun[r]);
        }
}

// ---------------------------------------------------------------------------
// Host orchestration
// ---------------------------------------------------------------------------
static void run_quant(const float* w, long n, unsigned short* wq, int tern,
                      float* partials, float* scale, hipStream_t s) {
    k_reduce_abs<<<256, 256, 0, s>>>(w, n, partials);
    k_finalize_scale<<<1, 256, 0, s>>>(partials, n, scale);
    long blocks = (n + 255) / 256; if (blocks > 2048) blocks = 2048;
    k_quant<<<(int)blocks, 256, 0, s>>>(w, n, scale, wq, tern);
}

extern "C" void kernel_launch(void* const* d_in, const int* in_sizes, int n_in,
                              void* d_out, int out_size, void* d_ws, size_t ws_size,
                              hipStream_t stream) {
    const float* x      = (const float*)d_in[0];
    const float* ln1_g  = (const float*)d_in[1];
    const float* ln1_b  = (const float*)d_in[2];
    const float* Wqkv   = (const float*)d_in[3];
    const float* ln2_g  = (const float*)d_in[4];
    const float* ln2_b  = (const float*)d_in[5];
    const float* Wo     = (const float*)d_in[6];
    const float* bo     = (const float*)d_in[7];
    const float* ff1_g  = (const float*)d_in[8];
    const float* ff1_b  = (const float*)d_in[9];
    const float* W1     = (const float*)d_in[10];
    const float* b1     = (const float*)d_in[11];
    const float* ff2_g  = (const float*)d_in[12];
    const float* ff2_b  = (const float*)d_in[13];
    const float* W2     = (const float*)d_in[14];
    const float* b2     = (const float*)d_in[15];
    (void)in_sizes; (void)n_in; (void)out_size; (void)ws_size;

    float* h = (float*)d_out;                              // residual stream, f32
    unsigned short* bufA = (unsigned short*)d_ws;          // [8192*3072] bf16
    unsigned short* bufB = bufA + (size_t)M_TOT * MLP_;    // [8192*3072] bf16
    unsigned short* wq   = bufB + (size_t)M_TOT * MLP_;    // [3072*768]  bf16
    float* partials = (float*)(wq + (size_t)MLP_ * DIM_);
    float* scale    = partials + 256;

    hipMemcpyAsync(h, x, sizeof(float) * (size_t)M_TOT * DIM_,
                   hipMemcpyDeviceToDevice, stream);

    for (int l = 0; l < 6; ++l) {
        const int tern = (l < 5);

        // ---- attention ----
        k_layernorm<<<M_TOT, 256, 0, stream>>>(h, nullptr, ln1_g + l * DIM_, ln1_b + l * DIM_,
                                               bufA, DIM_);
        run_quant(Wqkv + (size_t)l * QKV_LD * DIM_, (long)QKV_LD * DIM_, wq, tern,
                  partials, scale, stream);
        k_gemm_bf16<<<dim3(M_TOT / 128, QKV_LD / 128), 256, 0, stream>>>(
            bufA, DIM_, wq, DIM_, nullptr, bufB, QKV_LD, nullptr, DIM_, 0);

        k_attention<<<M_TOT * HEADS_ / 16 / 8, 256, 0, stream>>>(bufB, bufA);

        k_layernorm<<<M_TOT, 256, 0, stream>>>(nullptr, bufA, ln2_g + l * INNER_, ln2_b + l * INNER_,
                                               bufB, INNER_);
        run_quant(Wo + (size_t)l * DIM_ * INNER_, (long)DIM_ * INNER_, wq, tern,
                  partials, scale, stream);
        k_gemm_bf16<<<dim3(M_TOT / 128, DIM_ / 128), 256, 0, stream>>>(
            bufB, INNER_, wq, INNER_, bo + l * DIM_, nullptr, DIM_, h, INNER_, 2);

        // ---- MLP ----
        k_layernorm<<<M_TOT, 256, 0, stream>>>(h, nullptr, ff1_g + l * DIM_, ff1_b + l * DIM_,
                                               bufA, DIM_);
        run_quant(W1 + (size_t)l * MLP_ * DIM_, (long)MLP_ * DIM_, wq, tern,
                  partials, scale, stream);
        k_gemm_bf16<<<dim3(M_TOT / 128, MLP_ / 128), 256, 0, stream>>>(
            bufA, DIM_, wq, DIM_, b1 + l * MLP_, bufB, MLP_, nullptr, DIM_, 1);

        k_layernorm<<<M_TOT, 256, 0, stream>>>(nullptr, bufB, ff2_g + l * MLP_, ff2_b + l * MLP_,
                                               bufA, MLP_);
        run_quant(W2 + (size_t)l * DIM_ * MLP_, (long)DIM_ * MLP_, wq, tern,
                  partials, scale, stream);
        k_gemm_bf16<<<dim3(M_TOT / 128, DIM_ / 128), 256, 0, stream>>>(
            bufA, MLP_, wq, MLP_, b2 + l * DIM_, nullptr, DIM_, h, MLP_, 2);
    }
}
